// EnhancedSelfAttention_39977555591431
// MI455X (gfx1250) — matrix operand order repaired and checked by reading, not compiled, not run on hardware
//
#include <hip/hip_runtime.h>
#include <hip/hip_bf16.h>

// N=8, S=2048, D=1024.  Output (N,S,2D) fp32.
#define NB 8
#define SS 2048
#define DD 1024

typedef __attribute__((ext_vector_type(16))) _Float16 v16h;
typedef __attribute__((ext_vector_type(8)))  float    v8f;
typedef __attribute__((ext_vector_type(4))) unsigned int v4u;
typedef __attribute__((ext_vector_type(8))) int v8i_;
typedef __attribute__((ext_vector_type(4))) int v4i_;

#if __has_builtin(__builtin_amdgcn_tensor_load_to_lds)
#define USE_TDM 1
#else
#define USE_TDM 0
#endif
#if __has_builtin(__builtin_amdgcn_s_wait_tensorcnt)
#define HAS_WAIT_TENSOR 1
#else
#define HAS_WAIT_TENSOR 0
#endif

static __device__ inline v8f wmma_f16(v16h a, v16h b, v8f c) {
  // D = A(16x32) * B(32x16) + C, f32 accumulate
  return __builtin_amdgcn_wmma_f32_16x16x32_f16(false, a, false, b, (short)0, c,
                                                false, false);
}

// A-fragment (16x32 f16, M x K) from a row-major [16][stride] f16 tile.
// Contiguous k order used here (not the ISA table order): lanes 0-15 -> M=0..15 K=0..15, lanes 16-31 -> K=16..31.
static __device__ inline v16h load_a_frag(const _Float16* base, int stride) {
  const int lane = threadIdx.x & 31;
  const int m = lane & 15;
  const int khalf = (lane >> 4) * 16;
  const _Float16* row = base + m * stride;
  v16h f;
#pragma unroll
  for (int j = 0; j < 16; ++j) f[j] = row[khalf + j];
  return f;
}

// B-fragment (32x16 f16, K x N) where the source is stored "N-major":
// tile[n][k] row-major with row length `stride` (i.e. B[k][n] = tile[n][k]).
// Each lane (column n = lane&15) reads 16 contiguous f16 (32 B) from row n.
static __device__ inline v16h load_b_frag_rowN(const _Float16* base, size_t stride) {
  const int lane = threadIdx.x & 31;
  const int n = lane & 15;
  const int koff = (lane >> 4) * 16;
  const _Float16* row = base + (size_t)n * stride + koff;
  v16h f;
#pragma unroll
  for (int j = 0; j < 16; ++j) f[j] = row[j];
  return f;
}

// ---------------------------------------------------------------------------
// Kernel 1: xmax[s,d] = max_b x[b,s,d]
// ---------------------------------------------------------------------------
__global__ __launch_bounds__(256) void xmax_kernel(const float* __restrict__ x,
                                                   float* __restrict__ xmax) {
  size_t idx = (size_t)blockIdx.x * 256 + threadIdx.x;  // over S*D
  float mx = -1e30f;
#pragma unroll
  for (int b = 0; b < NB; ++b)
    mx = fmaxf(mx, x[(size_t)b * SS * DD + idx]);
  xmax[idx] = mx;
}

// ---------------------------------------------------------------------------
// Kernel 2: LDS-tiled staging.  For each (b, 32x32 tile of (s,d)):
//   - write passthrough out[b,s,0:D] = x              (coalesced along d)
//   - write xT16[b][d][s]   = f16(x)                  (coalesced along s)
//   - write lhsT16[b][d][s] = f16(xmax - x)           (coalesced along s)
// ---------------------------------------------------------------------------
__global__ __launch_bounds__(256) void stage_kernel(
    const float* __restrict__ x, const float* __restrict__ xmax,
    _Float16* __restrict__ xT16, _Float16* __restrict__ lhsT16,
    float* __restrict__ out) {
  __shared__ _Float16 tx[32 * 33];
  __shared__ _Float16 tl[32 * 33];
  const int tid = threadIdx.x;
  const int b = blockIdx.y;
  const int s0 = (blockIdx.x >> 5) * 32;  // 64 s-tiles
  const int d0 = (blockIdx.x & 31) * 32;  // 32 d-tiles

  {
    const int srow = tid >> 3;
    const int dcol = (tid & 7) * 4;
    const float4 xv = *(const float4*)(x + ((size_t)b * SS + s0 + srow) * DD + d0 + dcol);
    const float4 mv = *(const float4*)(xmax + (size_t)(s0 + srow) * DD + d0 + dcol);
    *(float4*)(out + ((size_t)b * SS + s0 + srow) * (2 * DD) + d0 + dcol) = xv;
    const float xa[4] = {xv.x, xv.y, xv.z, xv.w};
    const float ma[4] = {mv.x, mv.y, mv.z, mv.w};
#pragma unroll
    for (int j = 0; j < 4; ++j) {
      tx[srow * 33 + dcol + j] = (_Float16)xa[j];
      tl[srow * 33 + dcol + j] = (_Float16)(ma[j] - xa[j]);
    }
  }
  __syncthreads();
  {
    const int drow = tid >> 3;
    const int scol = (tid & 7) * 4;
    _Float16* xo = xT16 + ((size_t)b * DD + d0 + drow) * SS + s0 + scol;
    _Float16* lo = lhsT16 + ((size_t)b * DD + d0 + drow) * SS + s0 + scol;
#pragma unroll
    for (int j = 0; j < 4; ++j) {
      xo[j] = tx[(scol + j) * 33 + drow];
      lo[j] = tl[(scol + j) * 33 + drow];
    }
  }
}

// ---------------------------------------------------------------------------
// Kernel 3: Y = relu(Xeff @ W^T + bias) in f16, Xeff = X or (Xmax - X).
// Block tile 128x128, 8 waves; wave w owns rows [w*16,w*16+16) x 8 col tiles.
// ---------------------------------------------------------------------------
__global__ __launch_bounds__(256) void proj_gemm_kernel(
    const float* __restrict__ X, const float* __restrict__ Xmax,
    const float* __restrict__ W, const float* __restrict__ bias,
    _Float16* __restrict__ Y) {
  __shared__ _Float16 Xs[128 * 32];
  __shared__ _Float16 Ws[128 * 32];

  const int tid = threadIdx.x;
  const int wave = tid >> 5;
  const int lane = tid & 31;
  const int ln = lane & 15;
  const int hi = lane >> 4;
  const int row0 = blockIdx.x * 128;
  const int col0 = blockIdx.y * 128;

  v8f zero = {};
  v8f acc[8];
#pragma unroll
  for (int t = 0; t < 8; ++t) acc[t] = zero;

  const int lr = tid >> 1;        // 0..127
  const int lc = (tid & 1) * 16;  // 0 or 16

  for (int kb = 0; kb < DD / 32; ++kb) {
    const int k0 = kb * 32;
    {
      const int grow = row0 + lr;
      const float* src = X + (size_t)grow * DD + k0 + lc;
      _Float16* dst = Xs + lr * 32 + lc;
      if (Xmax) {
        const float* mx = Xmax + (size_t)(grow & (SS - 1)) * DD + k0 + lc;
#pragma unroll
        for (int j = 0; j < 16; ++j) dst[j] = (_Float16)(mx[j] - src[j]);
      } else {
#pragma unroll
        for (int j = 0; j < 16; ++j) dst[j] = (_Float16)src[j];
      }
    }
    {  // Ws[n][k] = W[(col0+n)*D + k0+k]
      const float* src = W + (size_t)(col0 + lr) * DD + k0 + lc;
      _Float16* dst = Ws + lr * 32 + lc;
#pragma unroll
      for (int j = 0; j < 16; ++j) dst[j] = (_Float16)src[j];
    }
    __syncthreads();
    v16h a = load_a_frag(Xs + wave * 16 * 32, 32);
#pragma unroll
    for (int t = 0; t < 8; ++t) {
      v16h b = load_b_frag_rowN(Ws + t * 16 * 32, 32);
      acc[t] = wmma_f16(a, b, acc[t]);
    }
    __syncthreads();
  }

#pragma unroll
  for (int t = 0; t < 8; ++t) {
    const int col = col0 + t * 16 + ln;
    const float bs = bias[col];
#pragma unroll
    for (int r = 0; r < 8; ++r) {
      const int row = row0 + wave * 16 + r + hi * 8;
      float v = acc[t][r] + bs;
      v = fmaxf(v, 0.0f);
      Y[(size_t)row * DD + col] = (_Float16)v;
    }
  }
}

// ---------------------------------------------------------------------------
// Kernel 4: flash attention  out[i,:] = softmax_j(Q_i.K_j / sqrt(D)) @ V
// Vt is V TRANSPOSED: [b][d][s] so PV B-fragments are contiguous per lane.
// Q tile (32 KB contiguous f16) pulled into LDS by the Tensor Data Mover
// (1-D descriptor, wave 0 only, TENSORcnt + workgroup barrier).
// MODE 0: out = coeff*o ; MODE 1: out += (1-coeff)*o
// ---------------------------------------------------------------------------
template <int MODE>
__global__ __launch_bounds__(256) void flash_attn_kernel(
    const _Float16* __restrict__ Q, const _Float16* __restrict__ K,
    const _Float16* __restrict__ Vt, const float* __restrict__ coeff,
    float* __restrict__ out) {
  __shared__ _Float16 Qs[16 * DD];   // 32 KB
  __shared__ _Float16 Ps[16 * 128];  // 4 KB
  __shared__ float m_sh[16], l_sh[16], alpha_sh[16];
  __shared__ float rmax_sh[8 * 16], rsum_sh[8 * 16];

  const int b = blockIdx.y;
  const int qb = blockIdx.x;
  const int tid = threadIdx.x;
  const int wave = tid >> 5;
  const int lane = tid & 31;
  const int ln = lane & 15;
  const int hi = lane >> 4;
  const float rscale = 0.03125f;  // 1/sqrt(1024)

  const _Float16* Qsrc = Q + ((size_t)b * SS + qb * 16) * DD;
#if USE_TDM
  if (wave == 0) {  // TDM: one 1-D tile of 16384 f16 elements -> LDS
    const unsigned lds_addr = (unsigned)(uintptr_t)&Qs[0];
    const unsigned long long ga = (unsigned long long)(uintptr_t)Qsrc;
    v4u g0;
    g0[0] = 1u;                                   // count=1, user mode
    g0[1] = lds_addr;                             // lds_addr[31:0]
    g0[2] = (unsigned)(ga & 0xffffffffu);         // global_addr[31:0]
    g0[3] = (unsigned)((ga >> 32) & 0x1ffffffu) | (2u << 30);  // addr[56:32],type=2
    v8i_ g1;
    g1[0] = 1 << 16;        // workgroup_mask=0, data_size=1 (2 bytes)
    g1[1] = 16384 << 16;    // tensor_dim0[15:0] in bits[63:48]  (=16384)
    g1[2] = 1 << 16;        // tensor_dim0[31:16]=0 | tensor_dim1=1
    g1[3] = 16384 << 16;    // tensor_dim1[31:16]=0 | tile_dim0=16384
    g1[4] = 1;              // tile_dim1=1, tile_dim2=0
    g1[5] = 16384;          // tensor_dim0_stride[31:0]
    g1[6] = 0;              // stride[47:32]=0, tensor_dim1_stride[15:0]=0
    g1[7] = 0;
    v4i_ gz = {};           // groups 2/3 unused (<=2-D descriptor)
    v8i_ gz8 = {};          // extended group (clang-23 form), unused
#if __clang_major__ >= 23
    __builtin_amdgcn_tensor_load_to_lds(g0, g1, gz, gz, gz8, 0);
#else
    __builtin_amdgcn_tensor_load_to_lds(g0, g1, gz, gz, 0);
#endif
#if HAS_WAIT_TENSOR
    __builtin_amdgcn_s_wait_tensorcnt(0);
#else
    asm volatile("s_wait_tensorcnt 0x0" ::: "memory");
#endif
  }
#else
  {  // fallback: cooperative vector copy
    const uint4* src = (const uint4*)Qsrc;
    uint4* dst = (uint4*)Qs;
    for (int i = tid; i < 16 * DD / 8; i += 256) dst[i] = src[i];
  }
#endif
  if (tid < 16) { m_sh[tid] = -1e30f; l_sh[tid] = 0.0f; }
  __syncthreads();

  v8f zero = {};
  v8f acc[8];
#pragma unroll
  for (int t = 0; t < 8; ++t) acc[t] = zero;

  for (int kb = 0; kb < SS / 128; ++kb) {
    // ---- scores for this wave's 16 keys ----
    const int jbase = kb * 128 + wave * 16;
    const _Float16* Kt = K + ((size_t)b * SS + jbase) * DD;
    v8f sc = zero;
#pragma unroll 4
    for (int kk = 0; kk < DD / 32; ++kk) {
      v16h a = load_a_frag(Qs + kk * 32, DD);
      v16h bb = load_b_frag_rowN(Kt + kk * 32, DD);
      sc = wmma_f16(a, bb, sc);
    }
    // ---- row max (butterfly over 16-lane halves) ----
#pragma unroll
    for (int r = 0; r < 8; ++r) {
      float v = sc[r] * rscale;
      sc[r] = v;
      v = fmaxf(v, __shfl_xor(v, 1, 32));
      v = fmaxf(v, __shfl_xor(v, 2, 32));
      v = fmaxf(v, __shfl_xor(v, 4, 32));
      v = fmaxf(v, __shfl_xor(v, 8, 32));
      if (ln == 0) rmax_sh[wave * 16 + r + hi * 8] = v;
    }
    __syncthreads();  // B0
    if (tid < 16) {
      const float m_old = m_sh[tid];
      float m_new = m_old;
#pragma unroll
      for (int w = 0; w < 8; ++w) m_new = fmaxf(m_new, rmax_sh[w * 16 + tid]);
      m_sh[tid] = m_new;
      alpha_sh[tid] = __expf(m_old - m_new);
    }
    __syncthreads();  // B1
    // ---- p = exp(s - m), stage P tile, row-sum partials ----
#pragma unroll
    for (int r = 0; r < 8; ++r) {
      const int row = r + hi * 8;
      float p = __expf(sc[r] - m_sh[row]);
      Ps[row * 128 + wave * 16 + ln] = (_Float16)p;
      float s = p;
      s += __shfl_xor(s, 1, 32);
      s += __shfl_xor(s, 2, 32);
      s += __shfl_xor(s, 4, 32);
      s += __shfl_xor(s, 8, 32);
      if (ln == 0) rsum_sh[wave * 16 + row] = s;
    }
    // ---- rescale accumulators ----
#pragma unroll
    for (int r = 0; r < 8; ++r) {
      const float al = alpha_sh[r + hi * 8];
#pragma unroll
      for (int t = 0; t < 8; ++t) acc[t][r] *= al;
    }
    __syncthreads();  // B2
    if (tid < 16) {
      float s = 0.0f;
#pragma unroll
      for (int w = 0; w < 8; ++w) s += rsum_sh[w * 16 + tid];
      l_sh[tid] = l_sh[tid] * alpha_sh[tid] + s;
    }
    // ---- PV: acc += P(16x128) @ V(128 x my 128 cols), V from V^T ----
#pragma unroll
    for (int c = 0; c < 4; ++c) {
      v16h a = load_a_frag(Ps + c * 32, 128);
#pragma unroll
      for (int t = 0; t < 8; ++t) {
        const _Float16* vb = Vt + ((size_t)b * DD + wave * 128 + t * 16) * SS +
                             kb * 128 + c * 32;
        v16h bb = load_b_frag_rowN(vb, SS);
        acc[t] = wmma_f16(a, bb, acc[t]);
      }
    }
    __syncthreads();  // B3
  }

  // ---- epilogue: normalize + coeff combine into out[..., D:2D] ----
  float* outp = out + ((size_t)(b * SS + qb * 16)) * (2 * DD) + DD;
#pragma unroll
  for (int t = 0; t < 8; ++t) {
    const int col = wave * 128 + t * 16 + ln;
    const float cf = coeff[col];
#pragma unroll
    for (int r = 0; r < 8; ++r) {
      const int row = r + hi * 8;
      const float val = acc[t][r] / l_sh[row];
      float* dst = outp + (size_t)row * (2 * DD) + col;
      if (MODE == 0)
        *dst = cf * val;
      else
        *dst = *dst + (1.0f - cf) * val;
    }
  }
}

// ---------------------------------------------------------------------------
extern "C" void kernel_launch(void* const* d_in, const int* in_sizes, int n_in,
                              void* d_out, int out_size, void* d_ws, size_t ws_size,
                              hipStream_t stream) {
  const float* x = (const float*)d_in[0];
  const float* W1 = (const float*)d_in[1];
  const float* b1 = (const float*)d_in[2];
  const float* W2 = (const float*)d_in[3];
  const float* b2 = (const float*)d_in[4];
  const float* coeff = (const float*)d_in[5];
  float* out = (float*)d_out;

  char* ws = (char*)d_ws;
  const size_t SZ = (size_t)NB * SS * DD * sizeof(_Float16);  // 32 MB
  _Float16* h16 = (_Float16*)(ws + 0 * SZ);
  _Float16* l16 = (_Float16*)(ws + 1 * SZ);
  _Float16* r16 = (_Float16*)(ws + 2 * SZ);
  _Float16* xT16 = (_Float16*)(ws + 3 * SZ);    // V^T for branch 1
  _Float16* lhsT16 = (_Float16*)(ws + 4 * SZ);  // V^T for branch 2
  float* xmax = (float*)(ws + 5 * SZ);          // 8 MB

  // 1. batch max
  xmax_kernel<<<(SS * DD) / 256, 256, 0, stream>>>(x, xmax);

  // 2. staging: passthrough half + transposed f16 V buffers
  stage_kernel<<<dim3((SS / 32) * (DD / 32), NB), 256, 0, stream>>>(
      x, xmax, xT16, lhsT16, out);

  // 3. projections: h = relu(x W1^T + b1), r = relu(x W2^T + b2),
  //                 l = relu((xmax - x) W2^T + b2)
  dim3 ggrid((NB * SS) / 128, DD / 128);
  proj_gemm_kernel<<<ggrid, 256, 0, stream>>>(x, nullptr, W1, b1, h16);
  proj_gemm_kernel<<<ggrid, 256, 0, stream>>>(x, nullptr, W2, b2, r16);
  proj_gemm_kernel<<<ggrid, 256, 0, stream>>>(x, xmax, W2, b2, l16);

  // 4. both attention branches, coeff-combine fused
  dim3 fgrid(SS / 16, NB);
  flash_attn_kernel<0><<<fgrid, 256, 0, stream>>>(h16, h16, xT16, coeff, out);
  flash_attn_kernel<1><<<fgrid, 256, 0, stream>>>(r16, l16, lhsT16, coeff, out);
}
